// AttnGruDecoder_54056458388051
// MI455X (gfx1250) — matrix operand order repaired and checked by reading, not compiled, not run on hardware
//
#include <hip/hip_runtime.h>
#include <stdint.h>

#define B_   32
#define T_   32
#define L_   128
#define E_   300
#define EP_  320      // E padded to multiple of 32 for WMMA K-loop
#define H_   1024
#define H3_  3072
#define V_   50000
#define START_ 1
#define NTB_ 391      // ceil(50000/128)

typedef __attribute__((ext_vector_type(16))) __bf16 v16bf;
typedef __attribute__((ext_vector_type(8)))  float  v8f;

struct Frag {
  union { v16bf v; uint4 q[2]; unsigned short s[16]; };
};

__device__ __forceinline__ unsigned short f2bf(float x) {
  unsigned u = __float_as_uint(x);
  return (unsigned short)((u + 0x7FFFu + ((u >> 16) & 1u)) >> 16);  // RNE
}
__device__ __forceinline__ float bf2f(unsigned short u) {
  return __uint_as_float(((unsigned)u) << 16);
}

// A-fragment (16x32 bf16, M x K): lane idx = row; 16 contiguous K per half (K=half*16+0..15)
__device__ __forceinline__ Frag ldA(const unsigned short* p, int stride, int lane) {
  int idx = lane & 15, half = lane >> 4;
  const unsigned short* r = p + (size_t)idx * stride + half * 16;
  Frag f;
  f.q[0] = *(const uint4*)(r);
  f.q[1] = *(const uint4*)(r + 8);
  return f;
}
// B-fragment (32x16 bf16, K x N) from row-major W[N][K]: lane idx = col n; 16 contiguous K per half
__device__ __forceinline__ Frag ldB(const unsigned short* p, int stride, int lane) {
  int idx = lane & 15, half = lane >> 4;
  const unsigned short* r = p + (size_t)idx * stride + half * 16;
  Frag f;
  f.q[0] = *(const uint4*)(r);
  f.q[1] = *(const uint4*)(r + 8);
  return f;
}
// B-fragment with on-the-fly fp32 -> bf16 conversion (fallback for streaming W_pred)
__device__ __forceinline__ Frag ldBf32(const float* p, int stride, int lane) {
  int idx = lane & 15, half = lane >> 4;
  const float* r = p + (size_t)idx * stride + half * 16;
  float4 a = *(const float4*)(r);
  float4 b = *(const float4*)(r + 4);
  float4 c = *(const float4*)(r + 8);
  float4 d = *(const float4*)(r + 12);
  Frag f;
  f.s[0]=f2bf(a.x);  f.s[1]=f2bf(a.y);  f.s[2]=f2bf(a.z);  f.s[3]=f2bf(a.w);
  f.s[4]=f2bf(b.x);  f.s[5]=f2bf(b.y);  f.s[6]=f2bf(b.z);  f.s[7]=f2bf(b.w);
  f.s[8]=f2bf(c.x);  f.s[9]=f2bf(c.y);  f.s[10]=f2bf(c.z); f.s[11]=f2bf(c.w);
  f.s[12]=f2bf(d.x); f.s[13]=f2bf(d.y); f.s[14]=f2bf(d.z); f.s[15]=f2bf(d.w);
  return f;
}
__device__ __forceinline__ Frag fragZero() {
  Frag f;
  f.q[0] = make_uint4(0,0,0,0);
  f.q[1] = make_uint4(0,0,0,0);
  return f;
}
__device__ __forceinline__ v8f wmma_bf16(const Frag& a, const Frag& b, v8f c) {
  return __builtin_amdgcn_wmma_f32_16x16x32_bf16(false, a.v, false, b.v, (short)0, c, false, false);
}
__device__ __forceinline__ float sigf(float x) { return 1.0f / (1.0f + __expf(-x)); }

// ---------------- prep kernels ----------------

__global__ void k_conv(const float* __restrict__ in, unsigned short* __restrict__ out, long n) {
  long i = (long)blockIdx.x * blockDim.x + threadIdx.x;
  long st = (long)gridDim.x * blockDim.x;
  for (; i < n; i += st) out[i] = f2bf(in[i]);
}

__global__ void k_wih(const float* __restrict__ w, unsigned short* __restrict__ out) {
  long n = (long)H3_ * EP_;
  long i = (long)blockIdx.x * blockDim.x + threadIdx.x;
  long st = (long)gridDim.x * blockDim.x;
  for (; i < n; i += st) {
    int e = (int)(i % EP_);
    long r = i / EP_;
    out[i] = (e < E_) ? f2bf(w[r * E_ + e]) : (unsigned short)0;
  }
}

__global__ void k_emb(const int* __restrict__ x, const float* __restrict__ embed,
                      unsigned short* __restrict__ out) {
  long n = (long)B_ * T_ * EP_;
  long i = (long)blockIdx.x * blockDim.x + threadIdx.x;
  long st = (long)gridDim.x * blockDim.x;
  for (; i < n; i += st) {
    int e = (int)(i % EP_);
    int r = (int)(i / EP_);
    int b = r / T_, t = r % T_;
    int tok = (t == 0) ? START_ : x[b * T_ + (t - 1)];
    out[i] = (e < E_) ? f2bf(embed[(size_t)tok * E_ + e]) : (unsigned short)0;
  }
}

__global__ void k_inith(const float* __restrict__ hs, float* __restrict__ hf,
                        unsigned short* __restrict__ hb) {
  int i = blockIdx.x * blockDim.x + threadIdx.x;
  if (i < B_ * H_) { hf[i] = hs[i]; hb[i] = f2bf(hs[i]); }
}

// gi_all[b*T+t][3H] = emb_bf @ w_ih_bf^T + b_ih ; M=1024 N=3072 K=320
__global__ void k_gi(const unsigned short* __restrict__ A, const unsigned short* __restrict__ Bw,
                     const float* __restrict__ b_ih, float* __restrict__ gi) {
  int lane = threadIdx.x & 31, wave = threadIdx.x >> 5;
  int bn = blockIdx.x % 24, bm = blockIdx.x / 24;
  int wm = wave >> 2, wn = wave & 3;
  int m_base = bm * 128 + wm * 64;
  int n_base = bn * 128 + wn * 32;
  v8f zero8 = {0,0,0,0,0,0,0,0};
  v8f acc[4][2];
#pragma unroll
  for (int mt = 0; mt < 4; mt++)
#pragma unroll
    for (int nt = 0; nt < 2; nt++) acc[mt][nt] = zero8;
  for (int k0 = 0; k0 < EP_; k0 += 32) {
    Frag af[4], bf[2];
#pragma unroll
    for (int mt = 0; mt < 4; mt++) af[mt] = ldA(A + (size_t)(m_base + mt*16) * EP_ + k0, EP_, lane);
#pragma unroll
    for (int nt = 0; nt < 2; nt++) bf[nt] = ldB(Bw + (size_t)(n_base + nt*16) * EP_ + k0, EP_, lane);
#pragma unroll
    for (int mt = 0; mt < 4; mt++)
#pragma unroll
      for (int nt = 0; nt < 2; nt++) acc[mt][nt] = wmma_bf16(af[mt], bf[nt], acc[mt][nt]);
  }
  int idx = lane & 15, half = lane >> 4;
#pragma unroll
  for (int mt = 0; mt < 4; mt++)
#pragma unroll
    for (int nt = 0; nt < 2; nt++) {
      int nn = n_base + nt * 16 + idx;
      float bias = b_ih[nn];
#pragma unroll
      for (int j = 0; j < 8; j++) {
        int m = m_base + mt * 16 + half * 8 + j;
        gi[(size_t)m * H3_ + nn] = acc[mt][nt][j] + bias;
      }
    }
}

// ---------------- per-step kernels ----------------

// S1: gh = h_bf @ w_hh^T (all 3 gates for a 16-col slab) + fused GRU elementwise -> hg
__global__ void k_s1(const unsigned short* __restrict__ h_bf, const float* __restrict__ h_f32,
                     const unsigned short* __restrict__ whh, const float* __restrict__ gi_all,
                     const float* __restrict__ b_hh, float* __restrict__ hg_f,
                     unsigned short* __restrict__ hg_b, int t) {
  int lane = threadIdx.x & 31, wave = threadIdx.x >> 5;
  int c0 = (blockIdx.x * 8 + wave) * 16;          // 64 n-tiles over H
  v8f zero8 = {0,0,0,0,0,0,0,0};
  v8f aR[2] = {zero8, zero8}, aZ[2] = {zero8, zero8}, aN[2] = {zero8, zero8};
  for (int k0 = 0; k0 < H_; k0 += 32) {
    Frag a0 = ldA(h_bf + k0, H_, lane);
    Frag a1 = ldA(h_bf + (size_t)16 * H_ + k0, H_, lane);
    Frag bR = ldB(whh + (size_t)(c0) * H_ + k0, H_, lane);
    Frag bZ = ldB(whh + (size_t)(H_ + c0) * H_ + k0, H_, lane);
    Frag bN = ldB(whh + (size_t)(2 * H_ + c0) * H_ + k0, H_, lane);
    aR[0] = wmma_bf16(a0, bR, aR[0]); aR[1] = wmma_bf16(a1, bR, aR[1]);
    aZ[0] = wmma_bf16(a0, bZ, aZ[0]); aZ[1] = wmma_bf16(a1, bZ, aZ[1]);
    aN[0] = wmma_bf16(a0, bN, aN[0]); aN[1] = wmma_bf16(a1, bN, aN[1]);
  }
  int idx = lane & 15, half = lane >> 4;
  int c = c0 + idx;
  float bhr = b_hh[c], bhz = b_hh[H_ + c], bhn = b_hh[2 * H_ + c];
#pragma unroll
  for (int mt = 0; mt < 2; mt++)
#pragma unroll
    for (int j = 0; j < 8; j++) {
      int b = mt * 16 + half * 8 + j;
      const float* gi = gi_all + (size_t)(b * T_ + t) * H3_;
      float r  = sigf(gi[c] + aR[mt][j] + bhr);
      float z  = sigf(gi[H_ + c] + aZ[mt][j] + bhz);
      float nn = tanhf(gi[2 * H_ + c] + r * (aN[mt][j] + bhn));
      float ho = h_f32[(size_t)b * H_ + c];
      float hn = (1.0f - z) * nn + z * ho;
      hg_f[(size_t)b * H_ + c] = hn;
      hg_b[(size_t)b * H_ + c] = f2bf(hn);
    }
}

// S2: q = hg @ W_att^T + b_att  (M=32, N=1024, K=1024)
__global__ void k_s2(const unsigned short* __restrict__ hg_b, const unsigned short* __restrict__ watt,
                     const float* __restrict__ b_att, float* __restrict__ q) {
  int lane = threadIdx.x & 31, wave = threadIdx.x >> 5;
  int c0 = (blockIdx.x * 8 + wave) * 16;
  v8f zero8 = {0,0,0,0,0,0,0,0};
  v8f acc[2] = {zero8, zero8};
  for (int k0 = 0; k0 < H_; k0 += 32) {
    Frag a0 = ldA(hg_b + k0, H_, lane);
    Frag a1 = ldA(hg_b + (size_t)16 * H_ + k0, H_, lane);
    Frag bb = ldB(watt + (size_t)c0 * H_ + k0, H_, lane);
    acc[0] = wmma_bf16(a0, bb, acc[0]);
    acc[1] = wmma_bf16(a1, bb, acc[1]);
  }
  int idx = lane & 15, half = lane >> 4;
  int c = c0 + idx;
  float bias = b_att[c];
#pragma unroll
  for (int mt = 0; mt < 2; mt++)
#pragma unroll
    for (int j = 0; j < 8; j++) {
      int b = mt * 16 + half * 8 + j;
      q[(size_t)b * H_ + c] = acc[mt][j] + bias;
    }
}

// S3: scores + softmax + ctx ; one workgroup per batch row
__global__ void k_s3(const float* __restrict__ q, const unsigned short* __restrict__ enc_b,
                     unsigned short* __restrict__ ctx_b) {
  __shared__ float q_lds[H_];
  __shared__ float red[256];
  __shared__ float sc[L_];
  __shared__ float att[L_];
  __shared__ float mx_s, inv_s;
  int b = blockIdx.x, tid = threadIdx.x;
  for (int i = tid; i < H_; i += 256) q_lds[i] = q[(size_t)b * H_ + i];
  __syncthreads();
  int l = tid & 127, kh = tid >> 7;
  const unsigned short* er = enc_b + ((size_t)(b * L_ + l)) * H_ + kh * 512;
  float p = 0.0f;
  for (int k = 0; k < 512; k += 8) {
    uint4 u = *(const uint4*)(er + k);
    const unsigned short* s = (const unsigned short*)&u;
#pragma unroll
    for (int i = 0; i < 8; i++) p += bf2f(s[i]) * q_lds[kh * 512 + k + i];
  }
  red[tid] = p;
  __syncthreads();
  if (tid < 128) sc[tid] = red[tid] + red[tid + 128];
  __syncthreads();
  if (tid == 0) {
    float m = -1e30f;
    for (int i = 0; i < L_; i++) m = fmaxf(m, sc[i]);
    float s = 0.0f;
    for (int i = 0; i < L_; i++) s += __expf(sc[i] - m);
    mx_s = m; inv_s = 1.0f / s;
  }
  __syncthreads();
  if (tid < 128) att[tid] = __expf(sc[tid] - mx_s) * inv_s;
  __syncthreads();
  float a0 = 0, a1 = 0, a2 = 0, a3 = 0;
  for (int l2 = 0; l2 < L_; l2++) {
    float a = att[l2];
    const unsigned short* e2 = enc_b + ((size_t)(b * L_ + l2)) * H_;
    a0 += a * bf2f(e2[tid]);
    a1 += a * bf2f(e2[tid + 256]);
    a2 += a * bf2f(e2[tid + 512]);
    a3 += a * bf2f(e2[tid + 768]);
  }
  ctx_b[(size_t)b * H_ + tid]       = f2bf(a0);
  ctx_b[(size_t)b * H_ + tid + 256] = f2bf(a1);
  ctx_b[(size_t)b * H_ + tid + 512] = f2bf(a2);
  ctx_b[(size_t)b * H_ + tid + 768] = f2bf(a3);
}

// S4: h = tanh([ctx,hg] @ W_ctx^T + b_ctx) -> carry + Hc row for final GEMM (M=32,N=1024,K=2048)
__global__ void k_s4(const unsigned short* __restrict__ ctx_b, const unsigned short* __restrict__ hg_b,
                     const unsigned short* __restrict__ wctx, const float* __restrict__ b_ctx,
                     float* __restrict__ h_f, unsigned short* __restrict__ h_b,
                     unsigned short* __restrict__ Hc, int t) {
  int lane = threadIdx.x & 31, wave = threadIdx.x >> 5;
  int c0 = (blockIdx.x * 8 + wave) * 16;
  v8f zero8 = {0,0,0,0,0,0,0,0};
  v8f acc[2] = {zero8, zero8};
  for (int k0 = 0; k0 < 2 * H_; k0 += 32) {
    const unsigned short* Asrc = (k0 < H_) ? ctx_b : hg_b;
    int kk = (k0 < H_) ? k0 : (k0 - H_);
    Frag a0 = ldA(Asrc + kk, H_, lane);
    Frag a1 = ldA(Asrc + (size_t)16 * H_ + kk, H_, lane);
    Frag bb = ldB(wctx + (size_t)c0 * (2 * H_) + k0, 2 * H_, lane);
    acc[0] = wmma_bf16(a0, bb, acc[0]);
    acc[1] = wmma_bf16(a1, bb, acc[1]);
  }
  int idx = lane & 15, half = lane >> 4;
  int c = c0 + idx;
  float bias = b_ctx[c];
#pragma unroll
  for (int mt = 0; mt < 2; mt++)
#pragma unroll
    for (int j = 0; j < 8; j++) {
      int b = mt * 16 + half * 8 + j;
      float v = tanhf(acc[mt][j] + bias);
      h_f[(size_t)b * H_ + c] = v;
      unsigned short vb = f2bf(v);
      h_b[(size_t)b * H_ + c] = vb;
      Hc[(size_t)(b * T_ + t) * H_ + c] = vb;
    }
}

// Final: logits = Hc @ W_pred^T + b_pred, bf16-weight path (W_pred pre-converted, L2-resident)
__global__ void k_logits_bf(const unsigned short* __restrict__ Hc, const unsigned short* __restrict__ Wpb,
                            const float* __restrict__ b_pred, float* __restrict__ out) {
  int lane = threadIdx.x & 31, wave = threadIdx.x >> 5;
  int bn = blockIdx.x % NTB_, bm = blockIdx.x / NTB_;
  int wm = wave >> 2, wn = wave & 3;
  int m_base = bm * 128 + wm * 64;
  int n_base = bn * 128 + wn * 32;
  v8f zero8 = {0,0,0,0,0,0,0,0};
  v8f acc[4][2];
#pragma unroll
  for (int mt = 0; mt < 4; mt++)
#pragma unroll
    for (int nt = 0; nt < 2; nt++) acc[mt][nt] = zero8;
  for (int k0 = 0; k0 < H_; k0 += 32) {
    Frag af[4], bf[2];
#pragma unroll
    for (int mt = 0; mt < 4; mt++) af[mt] = ldA(Hc + (size_t)(m_base + mt*16) * H_ + k0, H_, lane);
#pragma unroll
    for (int nt = 0; nt < 2; nt++) {
      int v0 = n_base + nt * 16;
      bf[nt] = (v0 < V_) ? ldB(Wpb + (size_t)v0 * H_ + k0, H_, lane) : fragZero();
    }
#pragma unroll
    for (int mt = 0; mt < 4; mt++)
#pragma unroll
      for (int nt = 0; nt < 2; nt++) acc[mt][nt] = wmma_bf16(af[mt], bf[nt], acc[mt][nt]);
  }
  int idx = lane & 15, half = lane >> 4;
#pragma unroll
  for (int mt = 0; mt < 4; mt++)
#pragma unroll
    for (int nt = 0; nt < 2; nt++) {
      int v = n_base + nt * 16 + idx;
      if (v < V_) {
        float bias = b_pred[v];
#pragma unroll
        for (int j = 0; j < 8; j++) {
          int m = m_base + mt * 16 + half * 8 + j;
          out[(size_t)m * V_ + v] = acc[mt][nt][j] + bias;
        }
      }
    }
}

// Fallback: logits with on-the-fly fp32 -> bf16 weight conversion (no big scratch needed)
__global__ void k_logits_f32(const unsigned short* __restrict__ Hc, const float* __restrict__ Wp,
                             const float* __restrict__ b_pred, float* __restrict__ out) {
  int lane = threadIdx.x & 31, wave = threadIdx.x >> 5;
  int bn = blockIdx.x % NTB_, bm = blockIdx.x / NTB_;
  int wm = wave >> 2, wn = wave & 3;
  int m_base = bm * 128 + wm * 64;
  int n_base = bn * 128 + wn * 32;
  v8f zero8 = {0,0,0,0,0,0,0,0};
  v8f acc[4][2];
#pragma unroll
  for (int mt = 0; mt < 4; mt++)
#pragma unroll
    for (int nt = 0; nt < 2; nt++) acc[mt][nt] = zero8;
  for (int k0 = 0; k0 < H_; k0 += 32) {
    Frag af[4], bf[2];
#pragma unroll
    for (int mt = 0; mt < 4; mt++) af[mt] = ldA(Hc + (size_t)(m_base + mt*16) * H_ + k0, H_, lane);
#pragma unroll
    for (int nt = 0; nt < 2; nt++) {
      int v0 = n_base + nt * 16;
      bf[nt] = (v0 < V_) ? ldBf32(Wp + (size_t)v0 * H_ + k0, H_, lane) : fragZero();
    }
#pragma unroll
    for (int mt = 0; mt < 4; mt++)
#pragma unroll
      for (int nt = 0; nt < 2; nt++) acc[mt][nt] = wmma_bf16(af[mt], bf[nt], acc[mt][nt]);
  }
  int idx = lane & 15, half = lane >> 4;
#pragma unroll
  for (int mt = 0; mt < 4; mt++)
#pragma unroll
    for (int nt = 0; nt < 2; nt++) {
      int v = n_base + nt * 16 + idx;
      if (v < V_) {
        float bias = b_pred[v];
#pragma unroll
        for (int j = 0; j < 8; j++) {
          int m = m_base + mt * 16 + half * 8 + j;
          out[(size_t)m * V_ + v] = acc[mt][nt][j] + bias;
        }
      }
    }
}

// ---------------- launch ----------------

extern "C" void kernel_launch(void* const* d_in, const int* in_sizes, int n_in,
                              void* d_out, int out_size, void* d_ws, size_t ws_size,
                              hipStream_t stream) {
  (void)in_sizes; (void)n_in; (void)out_size;
  const int*   x      = (const int*)d_in[0];
  const float* hidden = (const float*)d_in[1];
  const float* enc    = (const float*)d_in[2];
  const float* embed  = (const float*)d_in[3];
  const float* w_ih   = (const float*)d_in[4];
  const float* w_hh   = (const float*)d_in[5];
  const float* b_ih   = (const float*)d_in[6];
  const float* b_hh   = (const float*)d_in[7];
  const float* W_att  = (const float*)d_in[8];
  const float* b_att  = (const float*)d_in[9];
  const float* W_ctx  = (const float*)d_in[10];
  const float* b_ctx  = (const float*)d_in[11];
  const float* W_pred = (const float*)d_in[12];
  const float* b_pred = (const float*)d_in[13];
  float* out = (float*)d_out;

  const size_t needBase = (size_t)40 << 20;    // ~39 MB core scratch
  const size_t needFull = (size_t)143 << 20;   // + 102.4 MB bf16 W_pred
  if (ws_size < needBase) return;
  bool bigWs = (ws_size >= needFull);

  char* w = (char*)d_ws;
  auto carve = [&](size_t bytes) -> char* {
    char* p = w;
    w += (bytes + 255) & ~(size_t)255;
    return p;
  };
  unsigned short* whh_b  = (unsigned short*)carve((size_t)H3_ * H_ * 2);
  unsigned short* watt_b = (unsigned short*)carve((size_t)H_ * H_ * 2);
  unsigned short* wctx_b = (unsigned short*)carve((size_t)H_ * 2 * H_ * 2);
  unsigned short* enc_b  = (unsigned short*)carve((size_t)B_ * L_ * H_ * 2);
  unsigned short* wih_b  = (unsigned short*)carve((size_t)H3_ * EP_ * 2);
  unsigned short* emb_b  = (unsigned short*)carve((size_t)B_ * T_ * EP_ * 2);
  float*          gi_all = (float*)carve((size_t)B_ * T_ * H3_ * 4);
  float*          h_f    = (float*)carve((size_t)B_ * H_ * 4);
  unsigned short* h_b    = (unsigned short*)carve((size_t)B_ * H_ * 2);
  float*          hg_f   = (float*)carve((size_t)B_ * H_ * 4);
  unsigned short* hg_b   = (unsigned short*)carve((size_t)B_ * H_ * 2);
  float*          q_f    = (float*)carve((size_t)B_ * H_ * 4);
  unsigned short* ctx_b  = (unsigned short*)carve((size_t)B_ * H_ * 2);
  unsigned short* Hc_b   = (unsigned short*)carve((size_t)B_ * T_ * H_ * 2);
  unsigned short* wpred_b = bigWs ? (unsigned short*)carve((size_t)V_ * H_ * 2) : nullptr;

  // one-time prep
  k_conv<<<1024, 256, 0, stream>>>(w_hh, whh_b, (long)H3_ * H_);
  k_conv<<<1024, 256, 0, stream>>>(W_att, watt_b, (long)H_ * H_);
  k_conv<<<1024, 256, 0, stream>>>(W_ctx, wctx_b, (long)H_ * 2 * H_);
  k_conv<<<1024, 256, 0, stream>>>(enc, enc_b, (long)B_ * L_ * H_);
  k_wih<<<1024, 256, 0, stream>>>(w_ih, wih_b);
  k_emb<<<512, 256, 0, stream>>>(x, embed, emb_b);
  k_inith<<<(B_ * H_ + 255) / 256, 256, 0, stream>>>(hidden, h_f, h_b);
  k_gi<<<192, 256, 0, stream>>>(emb_b, wih_b, b_ih, gi_all);
  if (bigWs)
    k_conv<<<4096, 256, 0, stream>>>(W_pred, wpred_b, (long)V_ * H_);

  // sequential recurrence (4 stream-ordered kernels per step)
  for (int t = 0; t < T_; t++) {
    k_s1<<<8, 256, 0, stream>>>(h_b, h_f, whh_b, gi_all, b_hh, hg_f, hg_b, t);
    k_s2<<<8, 256, 0, stream>>>(hg_b, watt_b, b_att, q_f);
    k_s3<<<32, 256, 0, stream>>>(q_f, enc_b, ctx_b);
    k_s4<<<8, 256, 0, stream>>>(ctx_b, hg_b, wctx_b, b_ctx, h_f, h_b, Hc_b, t);
  }

  // big fused vocab projection
  if (bigWs)
    k_logits_bf<<<8 * NTB_, 256, 0, stream>>>(Hc_b, wpred_b, b_pred, out);
  else
    k_logits_f32<<<8 * NTB_, 256, 0, stream>>>(Hc_b, W_pred, b_pred, out);
}
